// Q_DimeNetPP_5952824672704
// MI455X (gfx1250) — matrix operand order repaired and checked by reading, not compiled, not run on hardware
//
#include <hip/hip_runtime.h>
#include <hip/hip_bf16.h>

// ---------------- problem constants (from reference) ----------------
#define N_NODES   20000
#define N_EDGES   400000
#define N_TRIP    1500000
#define N_BATCH   128
#define HID       128
#define INTD      64
#define OUTEMB    256
#define NRAD      6
#define SBFD      42
#define NLAY      4
#define NBS_      1
#define NAS_      2
#define NOUT_     3

typedef __attribute__((ext_vector_type(16))) _Float16 v16h;
typedef __attribute__((ext_vector_type(8)))  float    v8f;

#define KSTR 68  // LDS K-stride in halves (padding for bank spread)

// =====================================================================
// Generic fused WMMA GEMM:  C[M,Nc] = epi( Aeff[M,K] @ W[K,Nc] )
//   Aeff = A (AMODE 0) | A*A2 (AMODE 1) | A+A2 (AMODE 2)   [compile-time]
//   epi(c) = [ +bias ] -> [ silu ] -> [ +res ]
// Block: 256 threads (8 waves), 64 rows x Nc cols per block.
// Wave w: row-tile rt = w>>1, column group cg = w&1 (Nc/2 cols each).
// TPW = Nc/32 column tiles per wave (compile-time: 2, 4 or 8).
// K staged to LDS in 64-wide f16 chunks; vectorized b128 fast path for
// full tiles, guarded scalar path for ragged edges / small K.
// =====================================================================
template<int TPW, int AMODE>
__global__ __launch_bounds__(256)
void k_gemm(const float* __restrict__ A, const float* __restrict__ A2,
            const float* __restrict__ W, const float* __restrict__ bias,
            const float* __restrict__ res, float* __restrict__ C,
            long long M, int K, int act)
{
    constexpr int Nc = TPW * 32;

    __shared__ _Float16 Alds[64 * KSTR];
    __shared__ _Float16 Wlds[256 * KSTR];

    const int tid  = threadIdx.x;
    const int lane = tid & 31;
    const int wave = tid >> 5;
    const int rt   = wave >> 1;          // row tile 0..3
    const int cg   = wave & 1;           // column group 0..1
    const int hi   = (lane >> 4) & 1;    // half-wave select
    const int m    = lane & 15;
    const long long row0 = (long long)blockIdx.x * 64;
    const bool fullM = (row0 + 64 <= M);

    v8f acc[TPW];
    #pragma unroll
    for (int t = 0; t < TPW; ++t)
        #pragma unroll
        for (int j = 0; j < 8; ++j) acc[t][j] = 0.f;

    const int nchunks = (K + 63) >> 6;
    for (int kc = 0; kc < nchunks; ++kc) {
        const int kbase = kc << 6;
        const bool fullK = (kbase + 64 <= K);

        // ---- stage A chunk: 64 x 64 (f32 -> f16, fused A2, zero pad) ----
        if (fullM && fullK) {
            // fast path: float4 global loads, packed b64 LDS stores
            #pragma unroll
            for (int it = 0; it < 4; ++it) {
                int e4 = (tid + it * 256) << 2;     // 0..4092 step 4
                int mr = e4 >> 6;
                int kk = e4 & 63;
                const long long off = (row0 + mr) * (long long)K + (kbase + kk);
                float4 a4 = *(const float4*)(A + off);
                if (AMODE == 1) {
                    float4 b4 = *(const float4*)(A2 + off);
                    a4.x *= b4.x; a4.y *= b4.y; a4.z *= b4.z; a4.w *= b4.w;
                } else if (AMODE == 2) {
                    float4 b4 = *(const float4*)(A2 + off);
                    a4.x += b4.x; a4.y += b4.y; a4.z += b4.z; a4.w += b4.w;
                }
                union { _Float16 h[4]; unsigned long long q; } pk;
                pk.h[0] = (_Float16)a4.x; pk.h[1] = (_Float16)a4.y;
                pk.h[2] = (_Float16)a4.z; pk.h[3] = (_Float16)a4.w;
                *(unsigned long long*)(&Alds[mr * KSTR + kk]) = pk.q;
            }
        } else {
            #pragma unroll
            for (int it = 0; it < 16; ++it) {
                int idx = tid + it * 256;           // 0..4095
                int mr  = idx >> 6;
                int kk  = idx & 63;
                long long rr = row0 + mr;
                int kg = kbase + kk;
                float a = 0.f;
                if (rr < M && kg < K) {
                    a = A[rr * (long long)K + kg];
                    if      (AMODE == 1) a *= A2[rr * (long long)K + kg];
                    else if (AMODE == 2) a += A2[rr * (long long)K + kg];
                }
                Alds[mr * KSTR + kk] = (_Float16)a;
            }
        }
        // ---- stage W chunk: 64 x Nc, stored K-major per column ----
        if (fullK) {
            for (int idx = tid; idx < (Nc << 4); idx += 256) {   // quads
                int e4 = idx << 2;
                int kk = e4 / Nc;
                int n  = e4 - kk * Nc;
                float4 w4 = *(const float4*)(W + (long long)(kbase + kk) * Nc + n);
                Wlds[(n + 0) * KSTR + kk] = (_Float16)w4.x;
                Wlds[(n + 1) * KSTR + kk] = (_Float16)w4.y;
                Wlds[(n + 2) * KSTR + kk] = (_Float16)w4.z;
                Wlds[(n + 3) * KSTR + kk] = (_Float16)w4.w;
            }
        } else {
            for (int idx = tid; idx < (Nc << 6); idx += 256) {
                int kk = idx / Nc;
                int n  = idx - kk * Nc;
                int kg = kbase + kk;
                float w = (kg < K) ? W[(long long)kg * Nc + n] : 0.f;
                Wlds[n * KSTR + kk] = (_Float16)w;
            }
        }
        __syncthreads();

        // ---- compute: two K=32 WMMA steps per chunk ----
        #pragma unroll
        for (int k0 = 0; k0 < 64; k0 += 32) {
            // A fragment (16x32 f16) for this wave's row tile
            union { v16h v; unsigned u[8]; } af;
            #pragma unroll
            for (int e2 = 0; e2 < 8; ++e2) {
                int kk = k0 + ((e2 >= 4) ? 16 : 0) + (hi ? 8 : 0) + ((e2 & 3) << 1);
                af.u[e2] = *(const unsigned*)(&Alds[(rt * 16 + m) * KSTR + kk]);
            }
            #pragma unroll
            for (int t = 0; t < TPW; ++t) {
                int n = cg * (Nc / 2) + t * 16 + m;
                union { v16h v; unsigned u[8]; } bf;
                #pragma unroll
                for (int e2 = 0; e2 < 8; ++e2) {
                    int kk = k0 + ((e2 >= 4) ? 16 : 0) + (hi ? 8 : 0) + ((e2 & 3) << 1);
                    bf.u[e2] = *(const unsigned*)(&Wlds[n * KSTR + kk]);
                }
                acc[t] = __builtin_amdgcn_wmma_f32_16x16x32_f16(
                    false, af.v, false, bf.v, (short)0, acc[t], false, false);
            }
        }
        __syncthreads();
    }

    // ---- epilogue: bias -> silu -> residual -> store ----
    if (fullM) {
        #pragma unroll
        for (int t = 0; t < TPW; ++t) {
            int col = cg * (Nc / 2) + t * 16 + m;
            #pragma unroll
            for (int r = 0; r < 8; ++r) {
                long long row = row0 + rt * 16 + r + (hi ? 8 : 0);
                float c = acc[t][r];
                if (bias) c += bias[col];
                if (act)  c = c / (1.f + __expf(-c));
                if (res)  c += res[row * (long long)Nc + col];
                C[row * (long long)Nc + col] = c;
            }
        }
    } else {
        #pragma unroll
        for (int t = 0; t < TPW; ++t) {
            int col = cg * (Nc / 2) + t * 16 + m;
            #pragma unroll
            for (int r = 0; r < 8; ++r) {
                long long row = row0 + rt * 16 + r + (hi ? 8 : 0);
                if (row < M) {
                    float c = acc[t][r];
                    if (bias) c += bias[col];
                    if (act)  c = c / (1.f + __expf(-c));
                    if (res)  c += res[row * (long long)Nc + col];
                    C[row * (long long)Nc + col] = c;
                }
            }
        }
    }
}

// ---------------- helper kernels ----------------
__global__ void k_zero(float* __restrict__ p, long long n) {
    long long g = (long long)blockIdx.x * 256 + threadIdx.x;
    if (g < n) p[g] = 0.f;
}

// segment-sum scatter: out[idx[r], c] += src[r,c] (*mult[r,c])
__global__ void k_scatter(const float* __restrict__ src, const float* __restrict__ mult,
                          const int* __restrict__ idx, float* __restrict__ out,
                          long long rows, int cols) {
    long long g = (long long)blockIdx.x * 256 + threadIdx.x;
    if (g >= rows * (long long)cols) return;
    long long r = g / cols;
    int c = (int)(g - r * cols);
    float v = src[g];
    if (mult) v *= mult[g];
    atomicAdd(&out[(long long)idx[r] * cols + c], v);
}

// triplet interaction: agg[idx_ji[t], c] += xkj[idx_kj[t], c] * s[t, c]   (c < 64)
__global__ void k_triplet(const float* __restrict__ xkj, const float* __restrict__ s,
                          const int* __restrict__ ikj, const int* __restrict__ iji,
                          float* __restrict__ agg, long long T) {
    long long g = (long long)blockIdx.x * 256 + threadIdx.x;
    if (g >= T * 64) return;
    long long t = g >> 6;
    int c = (int)(g & 63);
    float v = xkj[(long long)ikj[t] * 64 + c] * s[t * 64 + c];
    atomicAdd(&agg[(long long)iji[t] * 64 + c], v);
}

// per-row dot with a [K,1] weight: out[r] = A[r,:] . w
__global__ void k_rowdot(const float* __restrict__ A, const float* __restrict__ w,
                         float* __restrict__ out, int rows, int K) {
    int r = blockIdx.x * 256 + threadIdx.x;
    if (r >= rows) return;
    float acc = 0.f;
    for (int k = 0; k < K; ++k) acc += A[(long long)r * K + k] * w[k];
    out[r] = acc;
}

// u[b, :] += su[b]   (broadcast [B,1] onto [B,H])
__global__ void k_bcast(float* __restrict__ u, const float* __restrict__ su, long long n) {
    long long g = (long long)blockIdx.x * 256 + threadIdx.x;
    if (g < n) u[g] += su[g >> 7];   // H = 128
}

// ---------------- host-side orchestration ----------------
static inline unsigned gblk(long long n, int b) { return (unsigned)((n + b - 1) / b); }

extern "C" void kernel_launch(void* const* d_in, const int* in_sizes, int n_in,
                              void* d_out, int out_size, void* d_ws, size_t ws_size,
                              hipStream_t stream) {
    // inputs in setup_inputs() order
    const float* x         = (const float*)d_in[0];
    const float* edge_attr = (const float*)d_in[1];
    const float* rbf       = (const float*)d_in[2];
    const float* sbf       = (const float*)d_in[3];
    const int*   iidx      = (const int*)d_in[4];
    const int*   idx_kj    = (const int*)d_in[5];
    const int*   idx_ji    = (const int*)d_in[6];
    const int*   batch     = (const int*)d_in[7];
    const float* W_node    = (const float*)d_in[8];
    const float* W_edge    = (const float*)d_in[9];
    const float* We_rbf1   = (const float*)d_in[10];
    const float* We_rbf2   = (const float*)d_in[11];
    const float* We_sbf1   = (const float*)d_in[12];
    const float* We_sbf2   = (const float*)d_in[13];
    const float* We_rbf    = (const float*)d_in[14];
    const float* We_kj     = (const float*)d_in[15];
    const float* be_kj     = (const float*)d_in[16];
    const float* We_ji     = (const float*)d_in[17];
    const float* be_ji     = (const float*)d_in[18];
    const float* We_down   = (const float*)d_in[19];
    const float* We_up     = (const float*)d_in[20];
    const float* We_cat    = (const float*)d_in[21];
    const float* Wrb       = (const float*)d_in[22];
    const float* brb       = (const float*)d_in[23];
    const float* We_lin    = (const float*)d_in[24];
    const float* be_lin    = (const float*)d_in[25];
    const float* Wra       = (const float*)d_in[26];
    const float* bra       = (const float*)d_in[27];
    const float* Wv_up     = (const float*)d_in[28];
    const float* bv_up     = (const float*)d_in[29];
    const float* Wv_lins   = (const float*)d_in[30];
    const float* bv_lins   = (const float*)d_in[31];
    const float* Wv_out    = (const float*)d_in[32];

    float* u = (float*)d_out;   // [B, H]

    // carve workspace
    float* p = (float*)d_ws;
    auto carve = [&](long long n) { float* r = p; p += n; return r; };
    const long long EH = (long long)N_EDGES * HID;
    const long long EI = (long long)N_EDGES * INTD;
    float* e1   = carve(EH);
    float* bufA = carve(EH);                       // x_ji
    float* bufB = carve(EH);                       // x_kj / e
    float* bufC = carve(EH);                       // r / h / rW
    float* xkj2 = carve(EI);
    float* aggb = carve(EI);
    float* sbuf = carve((long long)N_TRIP * INTD);
    float* vn0  = carve((long long)N_NODES * HID);
    float* vn1  = carve((long long)N_NODES * OUTEMB);
    float* vn2  = carve((long long)N_NODES * OUTEMB);
    float* vnO  = carve(N_NODES);
    float* Wr   = carve(NRAD * HID);
    float* Ws   = carve(SBFD * INTD);
    float* su   = carve(N_BATCH);

    auto gemm = [&](const float* A, const float* A2, int amode, const float* W,
                    const float* bias, const float* res, float* C,
                    long long M, int K, int Nc, int act) {
        unsigned g = gblk(M, 64);
        #define GEMM_DISPATCH(T)                                                        \
            if (amode == 0)      k_gemm<T,0><<<g,256,0,stream>>>(A,A2,W,bias,res,C,M,K,act); \
            else if (amode == 1) k_gemm<T,1><<<g,256,0,stream>>>(A,A2,W,bias,res,C,M,K,act); \
            else                 k_gemm<T,2><<<g,256,0,stream>>>(A,A2,W,bias,res,C,M,K,act);
        if (Nc == 64)       { GEMM_DISPATCH(2) }
        else if (Nc == 128) { GEMM_DISPATCH(4) }
        else                { GEMM_DISPATCH(8) }
        #undef GEMM_DISPATCH
    };
    auto zero = [&](float* ptr, long long n) {
        k_zero<<<gblk(n, 256), 256, 0, stream>>>(ptr, n);
    };

    // ---- init: u = segment_sum(x @ W_node, batch); e1 = edge_attr @ W_edge ----
    zero(u, (long long)N_BATCH * HID);
    gemm(x, nullptr, 0, W_node, nullptr, nullptr, vn0, N_NODES, 48, HID, 0);
    k_scatter<<<gblk((long long)N_NODES * HID, 256), 256, 0, stream>>>(
        vn0, nullptr, batch, u, N_NODES, HID);
    gemm(edge_attr, nullptr, 0, W_edge, nullptr, nullptr, e1, N_EDGES, 12, HID, 0);

    for (int l = 0; l < NLAY; ++l) {
        // ---- update_e ----
        gemm(e1, nullptr, 0, We_ji + (long long)l*HID*HID, be_ji + l*HID, nullptr,
             bufA, N_EDGES, HID, HID, 1);                               // x_ji
        gemm(e1, nullptr, 0, We_kj + (long long)l*HID*HID, be_kj + l*HID, nullptr,
             bufB, N_EDGES, HID, HID, 1);                               // x_kj
        // r = rbf @ (We_rbf1 @ We_rbf2)
        gemm(We_rbf1 + l*NRAD*8, nullptr, 0, We_rbf2 + l*8*HID, nullptr, nullptr,
             Wr, NRAD, 8, HID, 0);
        gemm(rbf, nullptr, 0, Wr, nullptr, nullptr, bufC, N_EDGES, NRAD, HID, 0);
        // x_kj = silu((x_kj * r) @ We_down)
        gemm(bufB, bufC, 1, We_down + (long long)l*HID*INTD, nullptr, nullptr,
             xkj2, N_EDGES, HID, INTD, 1);
        // s = sbf @ (We_sbf1 @ We_sbf2)
        gemm(We_sbf1 + l*SBFD*8, nullptr, 0, We_sbf2 + l*8*INTD, nullptr, nullptr,
             Ws, SBFD, 8, INTD, 0);
        gemm(sbf, nullptr, 0, Ws, nullptr, nullptr, sbuf, N_TRIP, SBFD, INTD, 0);
        // agg = segment_sum(x_kj[idx_kj] * s, idx_ji)
        zero(aggb, EI);
        k_triplet<<<gblk((long long)N_TRIP * 64, 256), 256, 0, stream>>>(
            xkj2, sbuf, idx_kj, idx_ji, aggb, N_TRIP);
        // x_kj = silu(agg @ We_up)
        gemm(aggb, nullptr, 0, We_up + (long long)l*INTD*HID, nullptr, nullptr,
             bufB, N_EDGES, INTD, HID, 1);
        // e = silu((x_ji + x_kj) @ We_cat)
        gemm(bufA, bufB, 2, We_cat + (long long)l*HID*HID, nullptr, nullptr,
             bufB, N_EDGES, HID, HID, 1);
        // residual before-skip (NBS=1)
        {
            const float* W0 = Wrb + (long long)l*2*HID*HID;
            const float* b0 = brb + l*2*HID;
            gemm(bufB, nullptr, 0, W0,           b0,        nullptr, bufC, N_EDGES, HID, HID, 1);
            gemm(bufC, nullptr, 0, W0 + HID*HID, b0 + HID,  bufB,    bufB, N_EDGES, HID, HID, 1);
        }
        // e = silu(e @ We_lin + b) + e1
        gemm(bufB, nullptr, 0, We_lin + (long long)l*HID*HID, be_lin + l*HID, e1,
             bufB, N_EDGES, HID, HID, 1);
        // residual after-skip (NAS=2)
        for (int r_ = 0; r_ < NAS_; ++r_) {
            const float* W0 = Wra + (long long)l*NAS_*2*HID*HID + (long long)r_*2*HID*HID;
            const float* b0 = bra + l*NAS_*2*HID + r_*2*HID;
            gemm(bufB, nullptr, 0, W0,           b0,        nullptr, bufC, N_EDGES, HID, HID, 1);
            gemm(bufC, nullptr, 0, W0 + HID*HID, b0 + HID,  bufB,    bufB, N_EDGES, HID, HID, 1);
        }
        { float* t = e1; e1 = bufB; bufB = t; }   // e1 <- e

        // ---- update_v: vn = segment_sum((rbf@We_rbf) * e1, i) ----
        gemm(rbf, nullptr, 0, We_rbf + (long long)l*NRAD*HID, nullptr, nullptr,
             bufC, N_EDGES, NRAD, HID, 0);
        zero(vn0, (long long)N_NODES * HID);
        k_scatter<<<gblk(EH, 256), 256, 0, stream>>>(e1, bufC, iidx, vn0, N_EDGES, HID);
        gemm(vn0, nullptr, 0, Wv_up + (long long)l*HID*OUTEMB, bv_up + l*OUTEMB, nullptr,
             vn1, N_NODES, HID, OUTEMB, 0);
        {
            const float* WL = Wv_lins + (long long)l*NOUT_*OUTEMB*OUTEMB;
            const float* bL = bv_lins + l*NOUT_*OUTEMB;
            gemm(vn1, nullptr, 0, WL,                     bL,            nullptr, vn2, N_NODES, OUTEMB, OUTEMB, 1);
            gemm(vn2, nullptr, 0, WL + 1LL*OUTEMB*OUTEMB, bL + OUTEMB,   nullptr, vn1, N_NODES, OUTEMB, OUTEMB, 1);
            gemm(vn1, nullptr, 0, WL + 2LL*OUTEMB*OUTEMB, bL + 2*OUTEMB, nullptr, vn2, N_NODES, OUTEMB, OUTEMB, 1);
        }
        k_rowdot<<<gblk(N_NODES, 256), 256, 0, stream>>>(vn2, Wv_out + l*OUTEMB, vnO, N_NODES, OUTEMB);
        // ---- update_u: u += broadcast(segment_sum(vn, batch)) ----
        zero(su, N_BATCH);
        k_scatter<<<gblk(N_NODES, 256), 256, 0, stream>>>(vnO, nullptr, batch, su, N_NODES, 1);
        k_bcast<<<gblk((long long)N_BATCH * HID, 256), 256, 0, stream>>>(u, su, (long long)N_BATCH * HID);
    }
    (void)in_sizes; (void)n_in; (void)out_size; (void)ws_size;
}
